// DLRM_13460427505961
// MI455X (gfx1250) — compile-verified, matrix operand order checked by reading, not run on hardware
//
#include <hip/hip_runtime.h>
#include <hip/hip_bf16.h>
#include <math.h>

// ---------------- problem constants (from reference) ----------------
#define BB    4096      // batch
#define TT    26        // tables
#define VV    200000    // rows per table
#define LL    20        // lookups per bag
#define DD    64        // emb dim
#define DENSE 13
#define TOPIN (DD * (1 + TT))   // 1728

typedef __attribute__((ext_vector_type(16))) __bf16   v16bf;
typedef __attribute__((ext_vector_type(8)))  float    v8f;
typedef __attribute__((ext_vector_type(4)))  unsigned uint4v;

union Frag { v16bf bf; uint4v u[2]; };

__device__ __forceinline__ unsigned short f2bfbits(float x) {
    unsigned u = __builtin_bit_cast(unsigned, x);
    unsigned r = u + 0x7fffu + ((u >> 16) & 1u);   // round-to-nearest-even
    return (unsigned short)(r >> 16);
}
__device__ __forceinline__ float bf2f(unsigned short b) {
    unsigned u = ((unsigned)b) << 16;
    return __builtin_bit_cast(float, u);
}

// ---------------- fp32 -> bf16 weight conversion ----------------
__global__ void cvt_bf16_kernel(const float* __restrict__ src,
                                unsigned short* __restrict__ dst, int n) {
    int i = blockIdx.x * blockDim.x + threadIdx.x;
    if (i < n) dst[i] = f2bfbits(src[i]);
}

// ---------------- bottom MLP layer 0: [B,13] -> relu -> [B,512] bf16 ----------------
__global__ void bottom0_kernel(const float* __restrict__ x,     // [B,13]
                               const float* __restrict__ w,     // [512,13]
                               const float* __restrict__ b,     // [512]
                               unsigned short* __restrict__ h0) // [B,512] bf16
{
    int i = blockIdx.x * blockDim.x + threadIdx.x;   // over B*512
    int bi = i >> 9, n = i & 511;
    const float* xr = x + bi * DENSE;
    const float* wr = w + n * DENSE;
    float s = b[n];
#pragma unroll
    for (int k = 0; k < DENSE; ++k) s += xr[k] * wr[k];
    h0[i] = f2bfbits(s > 0.f ? s : 0.f);
}

// ---------------- WMMA bf16 GEMM: C = relu(A @ W^T + bias) -------------------
// A: [M,K] bf16 (pitch lda), W: [N,K] bf16, C written bf16 at [row, ccol0+col].
// Block = 256 threads (8 waves); block tile 64x128; each wave computes a 16x64
// strip = 4 accumulators, reusing its A fragment 4x. No LDS, no barriers:
// weights are L2/L0-resident, A fragments are shared via WGP$, and the
// barrier-free loop lets the compiler pipeline global_load_b128 clauses
// across the unrolled k-iterations instead of draining counters per step.
__global__ void gemm_bf16_relu_kernel(const unsigned short* __restrict__ A, int lda,
                                      const unsigned short* __restrict__ W,
                                      const float* __restrict__ bias,
                                      unsigned short* __restrict__ C, int ldc, int ccol0,
                                      int N, int K)
{
    const int wave  = threadIdx.x >> 5;
    const int lane  = threadIdx.x & 31;
    const int m0    = blockIdx.x * 64 + (wave >> 1) * 16;   // 4 M-tiles per block
    const int nbase = blockIdx.y * 128 + (wave & 1) * 64;   // 2 N-strips per block
    if (nbase >= N) return;                                  // safe: no barriers

    const int row = lane & 15;          // A: M row / W: N row / D: column
    const int kb  = (lane >> 4) << 3;   // 0 or 8 (ISA 16-bit operand layout)

    const unsigned short* arow = A + (size_t)(m0 + row) * lda + kb;
    const unsigned short* wp0  = W + (size_t)(nbase + row) * K + kb;
    const unsigned short* wp1  = wp0 + (size_t)16 * K;
    const unsigned short* wp2  = wp0 + (size_t)32 * K;
    const unsigned short* wp3  = wp0 + (size_t)48 * K;

    v8f acc0 = {}, acc1 = {}, acc2 = {}, acc3 = {};

#pragma unroll 2
    for (int k0 = 0; k0 < K; k0 += 32) {
        Frag a, b0, b1, b2, b3;
        a.u[0]  = *(const uint4v*)(arow + k0);        // K = kb .. kb+7
        a.u[1]  = *(const uint4v*)(arow + k0 + 16);   // K = kb+16 .. kb+23
        b0.u[0] = *(const uint4v*)(wp0 + k0);
        b0.u[1] = *(const uint4v*)(wp0 + k0 + 16);
        b1.u[0] = *(const uint4v*)(wp1 + k0);
        b1.u[1] = *(const uint4v*)(wp1 + k0 + 16);
        b2.u[0] = *(const uint4v*)(wp2 + k0);
        b2.u[1] = *(const uint4v*)(wp2 + k0 + 16);
        b3.u[0] = *(const uint4v*)(wp3 + k0);
        b3.u[1] = *(const uint4v*)(wp3 + k0 + 16);
        acc0 = __builtin_amdgcn_wmma_f32_16x16x32_bf16(false, a.bf, false, b0.bf,
                                                       (short)0, acc0, false, false);
        acc1 = __builtin_amdgcn_wmma_f32_16x16x32_bf16(false, a.bf, false, b1.bf,
                                                       (short)0, acc1, false, false);
        acc2 = __builtin_amdgcn_wmma_f32_16x16x32_bf16(false, a.bf, false, b2.bf,
                                                       (short)0, acc2, false, false);
        acc3 = __builtin_amdgcn_wmma_f32_16x16x32_bf16(false, a.bf, false, b3.bf,
                                                       (short)0, acc3, false, false);
    }

    // epilogue: D layout -> col = lane&15 (fixed), rows = (lane>>4)*8 + i
    const int rbase = (lane >> 4) << 3;
    v8f accs[4] = {acc0, acc1, acc2, acc3};
#pragma unroll
    for (int j = 0; j < 4; ++j) {
        const int col = nbase + j * 16 + row;
        const float bn = bias[col];
#pragma unroll
        for (int i = 0; i < 8; ++i) {
            float v = accs[j][i] + bn;
            v = v > 0.f ? v : 0.f;
            C[(size_t)(m0 + rbase + i) * ldc + ccol0 + col] = f2bfbits(v);
        }
    }
}

// ---------------- embedding bag sum: one wave per (table, bag) ----------------
// Bandwidth-dominant part: each table row is 256B = one wave x float2/lane.
// All 20 bag indices are fetched with a single masked per-lane load (they are
// 80 contiguous bytes) and broadcast via shuffles; row l+2 is prefetched.
__global__ void embbag_kernel(const int* __restrict__ idx,     // [T,B,L]
                              const float* __restrict__ tables,// [T,V,D]
                              unsigned short* __restrict__ feat)// [B,TOPIN] bf16
{
    int w    = (blockIdx.x * blockDim.x + threadIdx.x) >> 5;   // global wave id
    int lane = threadIdx.x & 31;
    if (w >= TT * BB) return;
    int t = w / BB, b = w - t * BB;

    const int*   ip  = idx + ((size_t)t * BB + b) * LL;
    const float* tab = tables + (size_t)t * VV * DD + lane * 2;

    int myidx = (lane < LL) ? ip[lane] : 0;        // one load for all 20 indices

    float2 acc = make_float2(0.f, 0.f);
#pragma unroll
    for (int l = 0; l < LL; ++l) {
        int il = __shfl(myidx, l, 32);
        if (l + 2 < LL) {
            int ipre = __shfl(myidx, l + 2, 32);
            __builtin_prefetch(tab + (size_t)ipre * DD, 0, 0);
        }
        float2 v = *(const float2*)(tab + (size_t)il * DD);    // 8B/lane, 256B/row
        acc.x += v.x; acc.y += v.y;
    }
    unsigned pack = ((unsigned)f2bfbits(acc.y) << 16) | (unsigned)f2bfbits(acc.x);
    *(unsigned*)&feat[(size_t)b * TOPIN + DD + t * DD + lane * 2] = pack;
}

// ---------------- final layer: 256 -> 1 + sigmoid, one wave per batch row ----------------
__global__ void top_final_kernel(const unsigned short* __restrict__ z1, // [B,256] bf16
                                 const float* __restrict__ w2,          // [256]
                                 const float* __restrict__ b2,          // [1]
                                 float* __restrict__ out)               // [B]
{
    int w    = (blockIdx.x * blockDim.x + threadIdx.x) >> 5;
    int lane = threadIdx.x & 31;
    if (w >= BB) return;

    const unsigned short* zr = z1 + (size_t)w * 256 + lane * 8;
    uint4v zu = *(const uint4v*)zr;                 // 8 bf16 per lane
    const float* wr = w2 + lane * 8;

    float s = 0.f;
#pragma unroll
    for (int i = 0; i < 4; ++i) {
        unsigned u = zu[i];
        s += bf2f((unsigned short)(u & 0xffff)) * wr[2 * i];
        s += bf2f((unsigned short)(u >> 16))    * wr[2 * i + 1];
    }
#pragma unroll
    for (int d = 16; d > 0; d >>= 1) s += __shfl_down(s, d, 32);
    if (lane == 0) out[w] = 1.f / (1.f + __expf(-(s + b2[0])));
}

// ---------------- launcher ----------------
extern "C" void kernel_launch(void* const* d_in, const int* in_sizes, int n_in,
                              void* d_out, int out_size, void* d_ws, size_t ws_size,
                              hipStream_t stream)
{
    const float* x_dense = (const float*)d_in[0];
    // d_in[1] = x_offsets (int64) -- layout is fixed arange(B)*L, unused
    const int*   x_idx   = (const int*)d_in[2];
    const float* tables  = (const float*)d_in[3];
    const float* bw0 = (const float*)d_in[4];  const float* bb0 = (const float*)d_in[5];
    const float* bw1 = (const float*)d_in[6];  const float* bb1 = (const float*)d_in[7];
    const float* bw2 = (const float*)d_in[8];  const float* bb2 = (const float*)d_in[9];
    const float* tw0 = (const float*)d_in[10]; const float* tb0 = (const float*)d_in[11];
    const float* tw1 = (const float*)d_in[12]; const float* tb1 = (const float*)d_in[13];
    const float* tw2 = (const float*)d_in[14]; const float* tb2 = (const float*)d_in[15];
    float* out = (float*)d_out;

    // ---- workspace layout (bf16 buffers, 256B-aligned) ----
    char* ws = (char*)d_ws;
    size_t off = 0;
    auto alloc = [&](size_t bytes) { char* p = ws + off; off += (bytes + 255) & ~(size_t)255; return p; };
    unsigned short* feat = (unsigned short*)alloc((size_t)BB * TOPIN * 2); // [B,1728]
    unsigned short* h0   = (unsigned short*)alloc((size_t)BB * 512 * 2);
    unsigned short* h1   = (unsigned short*)alloc((size_t)BB * 256 * 2);
    unsigned short* z0   = (unsigned short*)alloc((size_t)BB * 512 * 2);
    unsigned short* z1   = (unsigned short*)alloc((size_t)BB * 256 * 2);
    unsigned short* bw1b = (unsigned short*)alloc((size_t)256 * 512 * 2);
    unsigned short* bw2b = (unsigned short*)alloc((size_t)64 * 256 * 2);
    unsigned short* tw0b = (unsigned short*)alloc((size_t)512 * TOPIN * 2);
    unsigned short* tw1b = (unsigned short*)alloc((size_t)256 * 512 * 2);

    // 1) weight conversions f32 -> bf16
    auto cvt = [&](const float* s, unsigned short* d, int n) {
        cvt_bf16_kernel<<<(n + 255) / 256, 256, 0, stream>>>(s, d, n);
    };
    cvt(bw1, bw1b, 256 * 512);
    cvt(bw2, bw2b, 64 * 256);
    cvt(tw0, tw0b, 512 * TOPIN);
    cvt(tw1, tw1b, 256 * 512);

    // 2) bottom MLP layer 0 (K=13, VALU)
    bottom0_kernel<<<(BB * 512) / 256, 256, 0, stream>>>(x_dense, bw0, bb0, h0);

    // 3) bottom layers 1,2 (WMMA): 512->256 then 256->64 (into feat[:,0:64])
    gemm_bf16_relu_kernel<<<dim3(BB / 64, 2), 256, 0, stream>>>(
        h0, 512, bw1b, bb1, h1, 256, 0, 256, 512);
    gemm_bf16_relu_kernel<<<dim3(BB / 64, 1), 256, 0, stream>>>(
        h1, 256, bw2b, bb2, feat, TOPIN, 0, 64, 256);

    // 4) embedding bags -> feat[:,64:1728] (bandwidth bound; 1 wave per bag)
    {
        int waves = TT * BB;                 // 106496
        int blocks = (waves * 32) / 256;     // 13312
        embbag_kernel<<<blocks, 256, 0, stream>>>(x_idx, tables, feat);
    }

    // 5) top MLP (WMMA): 1728->512, 512->256
    gemm_bf16_relu_kernel<<<dim3(BB / 64, 4), 256, 0, stream>>>(
        feat, TOPIN, tw0b, tb0, z0, 512, 0, 512, TOPIN);
    gemm_bf16_relu_kernel<<<dim3(BB / 64, 2), 256, 0, stream>>>(
        z0, 512, tw1b, tb1, z1, 256, 0, 256, 512);

    // 6) final 256->1 dot + sigmoid
    top_final_kernel<<<(BB * 32) / 256, 256, 0, stream>>>(z1, tw2, tb2, out);
}
